// LlamaForCausalLMWarpper_28681791603340
// MI455X (gfx1250) — compile-verified, matrix operand order checked by reading, not run on hardware
//
#include <hip/hip_runtime.h>
#include <hip/hip_bf16.h>

typedef __attribute__((ext_vector_type(16))) __bf16 v16bf;
typedef __attribute__((ext_vector_type(8)))  float  v8f;

#define DIM_H     4096
#define DIM_V     32000
#define NROWS     4096            // B*S = 2*2048
#define M_TILE    128
#define N_TILE    128
#define K_TILE    32
#define NSPLIT    10
#define COLS_PER_SPLIT (DIM_V / NSPLIT)     // 3200
#define N_ITERS   (COLS_PER_SPLIT / N_TILE) // 25
#define K_ITERS   (DIM_H / K_TILE)          // 128
#define IGNORE_INDEX (-100)
#define NEG_LARGE (-3.0e38f)

union Frag {
    uint4 q[2];
    v16bf v;
};

__device__ __forceinline__ unsigned int pkbf(float a, float b) {
    union { __bf16 h[2]; unsigned int u; } r;
    r.h[0] = (__bf16)a;
    r.h[1] = (__bf16)b;
    return r.u;
}

__device__ __forceinline__ void store_slab(unsigned int* p,
                                           float4 a0, float4 a1,
                                           float4 a2, float4 a3) {
    p[0] = pkbf(a0.x, a0.y); p[1] = pkbf(a0.z, a0.w);
    p[2] = pkbf(a1.x, a1.y); p[3] = pkbf(a1.z, a1.w);
    p[4] = pkbf(a2.x, a2.y); p[5] = pkbf(a2.z, a2.w);
    p[6] = pkbf(a3.x, a3.y); p[7] = pkbf(a3.z, a3.w);
}

// ---------------------------------------------------------------------------
// Kernel 1: tiled bf16 WMMA GEMM fused with online softmax statistics.
// grid = (NROWS/M_TILE, NSPLIT), block = 256 (8 waves of 32).
// Wave w owns C-rows [w*16, w*16+16) of the 128x128 tile.
// LDS is double-buffered; B fragments are register-ping-ponged so ds_load
// of fragment g+1 overlaps the WMMA of fragment g.
// ---------------------------------------------------------------------------
__global__ __launch_bounds__(256)
void lmhead_ce_partial(const float* __restrict__ hs,
                       const int*   __restrict__ labels,
                       const float* __restrict__ w,
                       float* __restrict__ pm,
                       float* __restrict__ ps,
                       float* __restrict__ pl) {
    __shared__ __align__(16) unsigned int lds_a[2][M_TILE * 16]; // 128 x 32 bf16
    __shared__ __align__(16) unsigned int lds_b[2][N_TILE * 16]; // 128 x 32 bf16

    const int tid  = threadIdx.x;
    const int wid  = tid >> 5;
    const int lane = tid & 31;
    const int half = lane >> 4;
    const int l15  = lane & 15;

    const int row0     = blockIdx.x * M_TILE;
    const int split    = blockIdx.y;
    const int col_base = split * COLS_PER_SPLIT;

    // Label target for each of the 8 C-rows this lane touches.
    // C layout: VGPR r -> M=r (lanes 0-15), M=8+r (lanes 16-31).
    int yr[8];
#pragma unroll
    for (int r = 0; r < 8; ++r)
        yr[r] = labels[row0 + wid * 16 + half * 8 + r];

    float m_run[8], s_run[8], ll[8];
#pragma unroll
    for (int r = 0; r < 8; ++r) { m_run[r] = NEG_LARGE; s_run[r] = 0.f; ll[r] = NEG_LARGE; }

    const int ld_r = tid >> 1;   // row (A) / col (B) staged by this thread
    const int ld_h = tid & 1;    // which 16-float half of the 32-wide K slab

    // Per-thread global staging bases (K advances by K_TILE floats).
    const float* gpa = hs + (size_t)(row0 + ld_r) * DIM_H + ld_h * 16;
    // Fragment LDS offsets (in uints) within one buffer.
    const int a_off0 = (wid * 16 + l15) * 16 + half * 4;
    const int a_off1 = a_off0 + 8;
    const int st_off = ld_r * 16 + ld_h * 8;

    for (int it = 0; it < N_ITERS; ++it) {
        const int col0 = col_base + it * N_TILE;
        const float* gpb = w + (size_t)(col0 + ld_r) * DIM_H + ld_h * 16;

        v8f acc[8];
#pragma unroll
        for (int g = 0; g < 8; ++g) {
            v8f z = {0.f, 0.f, 0.f, 0.f, 0.f, 0.f, 0.f, 0.f};
            acc[g] = z;
        }

        // ---- prologue: stage K-slab 0 into buffer 0 ----
        {
            const float4* ga = (const float4*)gpa;
            const float4* gb = (const float4*)gpb;
            float4 a0 = ga[0], a1 = ga[1], a2 = ga[2], a3 = ga[3];
            float4 b0 = gb[0], b1 = gb[1], b2 = gb[2], b3 = gb[3];
            store_slab(&lds_a[0][st_off], a0, a1, a2, a3);
            store_slab(&lds_b[0][st_off], b0, b1, b2, b3);
        }
        __syncthreads();

        for (int kt = 0; kt < K_ITERS; ++kt) {
            const int cur = kt & 1;
            const bool have_next = (kt + 1 < K_ITERS);

            // Issue next K-slab global loads early; they overlap the WMMAs.
            float4 a0, a1, a2, a3, b0, b1, b2, b3;
            if (have_next) {
                const float4* ga = (const float4*)(gpa + (size_t)(kt + 1) * K_TILE);
                const float4* gb = (const float4*)(gpb + (size_t)(kt + 1) * K_TILE);
                a0 = ga[0]; a1 = ga[1]; a2 = ga[2]; a3 = ga[3];
                b0 = gb[0]; b1 = gb[1]; b2 = gb[2]; b3 = gb[3];
                if (kt + 2 < K_ITERS)
                    __builtin_prefetch((const void*)(gb + 8), 0, 0);
            }

            // A fragment (16x32 bf16): lane holds row wid*16 + l15;
            // half 0 -> K{0..7,16..23}, half 1 -> K{8..15,24..31}.
            Frag fa;
            fa.q[0] = *(const uint4*)&lds_a[cur][a_off0];
            fa.q[1] = *(const uint4*)&lds_a[cur][a_off1];

            // Ping-pong B fragments so ds_load(g+1) overlaps wmma(g).
            Frag fb[2];
            {
                const int bc = l15 * 16 + half * 4;
                fb[0].q[0] = *(const uint4*)&lds_b[cur][bc];
                fb[0].q[1] = *(const uint4*)&lds_b[cur][bc + 8];
            }
#pragma unroll
            for (int g = 0; g < 8; ++g) {
                if (g < 7) {
                    const int bc = ((g + 1) * 16 + l15) * 16 + half * 4;
                    fb[(g + 1) & 1].q[0] = *(const uint4*)&lds_b[cur][bc];
                    fb[(g + 1) & 1].q[1] = *(const uint4*)&lds_b[cur][bc + 8];
                }
                acc[g] = __builtin_amdgcn_wmma_f32_16x16x32_bf16(
                    false, fa.v, false, fb[g & 1].v, (short)0, acc[g], false, false);
            }

            // Stage next slab into the other buffer, then one barrier.
            if (have_next) {
                store_slab(&lds_a[cur ^ 1][st_off], a0, a1, a2, a3);
                store_slab(&lds_b[cur ^ 1][st_off], b0, b1, b2, b3);
            }
            __syncthreads();
        }

        // ---- online softmax update over this 128-column slab ----
        float tmax[8];
#pragma unroll
        for (int r = 0; r < 8; ++r) tmax[r] = NEG_LARGE;
#pragma unroll
        for (int g = 0; g < 8; ++g)
#pragma unroll
            for (int r = 0; r < 8; ++r) tmax[r] = fmaxf(tmax[r], acc[g][r]);
#pragma unroll
        for (int r = 0; r < 8; ++r) {
            tmax[r] = fmaxf(tmax[r], __shfl_xor(tmax[r], 1, 32));
            tmax[r] = fmaxf(tmax[r], __shfl_xor(tmax[r], 2, 32));
            tmax[r] = fmaxf(tmax[r], __shfl_xor(tmax[r], 4, 32));
            tmax[r] = fmaxf(tmax[r], __shfl_xor(tmax[r], 8, 32));
        }

        float se[8];
#pragma unroll
        for (int r = 0; r < 8; ++r) se[r] = 0.f;
#pragma unroll
        for (int g = 0; g < 8; ++g) {
            const int col = col0 + g * 16 + l15;
#pragma unroll
            for (int r = 0; r < 8; ++r) {
                const float v = acc[g][r];
                se[r] += __expf(v - tmax[r]);
                if (col == yr[r]) ll[r] = v;
            }
        }
#pragma unroll
        for (int r = 0; r < 8; ++r) {
            se[r] += __shfl_xor(se[r], 1, 32);
            se[r] += __shfl_xor(se[r], 2, 32);
            se[r] += __shfl_xor(se[r], 4, 32);
            se[r] += __shfl_xor(se[r], 8, 32);
        }
#pragma unroll
        for (int r = 0; r < 8; ++r) {
            const float mn = fmaxf(m_run[r], tmax[r]);
            s_run[r] = s_run[r] * __expf(m_run[r] - mn) + se[r] * __expf(tmax[r] - mn);
            m_run[r] = mn;
        }
    }

    // Combine label logit across the 16 column-lanes.
#pragma unroll
    for (int r = 0; r < 8; ++r) {
        ll[r] = fmaxf(ll[r], __shfl_xor(ll[r], 1, 32));
        ll[r] = fmaxf(ll[r], __shfl_xor(ll[r], 2, 32));
        ll[r] = fmaxf(ll[r], __shfl_xor(ll[r], 4, 32));
        ll[r] = fmaxf(ll[r], __shfl_xor(ll[r], 8, 32));
    }

    if (l15 == 0) {
#pragma unroll
        for (int r = 0; r < 8; ++r) {
            const int row = row0 + wid * 16 + half * 8 + r;
            const size_t idx = (size_t)row * NSPLIT + split;
            pm[idx] = m_run[r];
            ps[idx] = s_run[r];
            pl[idx] = ll[r];
        }
    }
}

// ---------------------------------------------------------------------------
// Kernel 2: per-row combine of split partials -> NLL, accumulate loss & count.
// ---------------------------------------------------------------------------
__global__ void lmhead_ce_reduce(const int* __restrict__ labels,
                                 const float* __restrict__ pm,
                                 const float* __restrict__ ps,
                                 const float* __restrict__ pl,
                                 float* __restrict__ accum) {
    const int row = blockIdx.x * blockDim.x + threadIdx.x;
    if (row >= NROWS) return;

    float m = NEG_LARGE;
#pragma unroll
    for (int i = 0; i < NSPLIT; ++i) m = fmaxf(m, pm[(size_t)row * NSPLIT + i]);
    float s = 0.f;
#pragma unroll
    for (int i = 0; i < NSPLIT; ++i)
        s += ps[(size_t)row * NSPLIT + i] * __expf(pm[(size_t)row * NSPLIT + i] - m);
    const float lse = m + __logf(s);

    float l = NEG_LARGE;
#pragma unroll
    for (int i = 0; i < NSPLIT; ++i) l = fmaxf(l, pl[(size_t)row * NSPLIT + i]);

    if (labels[row] != IGNORE_INDEX) {
        atomicAdd(&accum[0], lse - l);
        atomicAdd(&accum[1], 1.0f);
    }
}

__global__ void lmhead_ce_init(float* __restrict__ accum) {
    accum[0] = 0.f;
    accum[1] = 0.f;
}

__global__ void lmhead_ce_final(const float* __restrict__ accum,
                                float* __restrict__ out) {
    out[0] = accum[0] / fmaxf(accum[1], 1.0f);
}

// ---------------------------------------------------------------------------
extern "C" void kernel_launch(void* const* d_in, const int* in_sizes, int n_in,
                              void* d_out, int out_size, void* d_ws, size_t ws_size,
                              hipStream_t stream) {
    const float* hs     = (const float*)d_in[0];   // [2,2048,4096] f32
    const int*   labels = (const int*)d_in[1];     // [2,2048]
    const float* w      = (const float*)d_in[2];   // [32000,4096] f32
    float*       out    = (float*)d_out;

    float* wsf = (float*)d_ws;
    float* pm    = wsf;                              // NROWS*NSPLIT
    float* ps    = wsf + (size_t)NROWS * NSPLIT;     // NROWS*NSPLIT
    float* pl    = wsf + (size_t)2 * NROWS * NSPLIT; // NROWS*NSPLIT
    float* accum = wsf + (size_t)3 * NROWS * NSPLIT; // 2 floats

    lmhead_ce_init<<<1, 1, 0, stream>>>(accum);

    dim3 grid(NROWS / M_TILE, NSPLIT);
    lmhead_ce_partial<<<grid, 256, 0, stream>>>(hs, labels, w, pm, ps, pl);

    lmhead_ce_reduce<<<NROWS / 256, 256, 0, stream>>>(labels, pm, ps, pl, accum);

    lmhead_ce_final<<<1, 1, 0, stream>>>(accum, out);
}
